// FusedINLLayer_40458591929054
// MI455X (gfx1250) — matrix operand order repaired and checked by reading, not compiled, not run on hardware
//
#include <hip/hip_runtime.h>

typedef __bf16 bf16;
typedef __attribute__((ext_vector_type(16))) __bf16 v16bf;
typedef __attribute__((ext_vector_type(8)))  float  v8f;

#define B_    4
#define S_    1024
#define DM_   1024
#define H_    16
#define HD_   64
#define FF_   4096
#define E_    4
#define KTOP_ 2
#define IT_   2
#define NT_   (B_*S_)   // 4096 tokens

#define BSTRIDE 40                     // padded LDS stride (halfwords) per column
#define BUFBYTES (64 * BSTRIDE * 2)    // one B-tile buffer in bytes

// ---------------------------------------------------------------------------
// CDNA5 async global->LDS copy (ASYNCcnt-tracked), GV addressing mode
// ---------------------------------------------------------------------------
__device__ __forceinline__ void async_copy_b128(unsigned ldsOff, const void* gptr) {
    asm volatile("global_load_async_to_lds_b128 %0, %1, off"
                 :: "v"(ldsOff), "v"(gptr) : "memory");
}
__device__ __forceinline__ void wait_async_le1() {
    asm volatile("s_wait_asynccnt 0x1" ::: "memory");
}
__device__ __forceinline__ void wait_async_le0() {
    asm volatile("s_wait_asynccnt 0x0" ::: "memory");
}

// ---------------------------------------------------------------------------
// Utility: f32 -> bf16 convert (grid-stride)
// ---------------------------------------------------------------------------
__global__ void cvt_bf16_kernel(const float* __restrict__ x, bf16* __restrict__ y, long n) {
    for (long i = (long)blockIdx.x * blockDim.x + threadIdx.x; i < n;
         i += (long)gridDim.x * blockDim.x)
        y[i] = (bf16)x[i];
}

// f32 [K,N] row-major -> bf16 [N,K] (transposed, so WMMA B-fragments are
// K-contiguous and load as b128)
__global__ void transpose_cvt_kernel(const float* __restrict__ W, bf16* __restrict__ Wt,
                                     int K, int N) {
    long total = (long)K * N;
    for (long i = (long)blockIdx.x * blockDim.x + threadIdx.x; i < total;
         i += (long)gridDim.x * blockDim.x) {
        int k = (int)(i / N);
        int n = (int)(i % N);
        Wt[(long)n * K + k] = (bf16)W[i];
    }
}

__global__ void copy_f32_kernel(const float* __restrict__ x, float* __restrict__ y, long n) {
    for (long i = (long)blockIdx.x * blockDim.x + threadIdx.x; i < n;
         i += (long)gridDim.x * blockDim.x)
        y[i] = x[i];
}

// ---------------------------------------------------------------------------
// LayerNorm over rows of DM_; writes bf16 (always) and f32 (optional)
// ---------------------------------------------------------------------------
__global__ __launch_bounds__(256) void ln_kernel(const float* __restrict__ x,
                                                 const float* __restrict__ g,
                                                 const float* __restrict__ b,
                                                 bf16* __restrict__ yb,
                                                 float* __restrict__ yf) {
    __shared__ float red[256];
    const int t = threadIdx.x;
    const float* xr = x + (long)blockIdx.x * DM_;
    float s = 0.f;
    for (int i = t; i < DM_; i += 256) s += xr[i];
    red[t] = s; __syncthreads();
    for (int st = 128; st > 0; st >>= 1) {
        if (t < st) red[t] += red[t + st];
        __syncthreads();
    }
    const float mean = red[0] * (1.0f / DM_);
    __syncthreads();
    float v = 0.f;
    for (int i = t; i < DM_; i += 256) { float d = xr[i] - mean; v += d * d; }
    red[t] = v; __syncthreads();
    for (int st = 128; st > 0; st >>= 1) {
        if (t < st) red[t] += red[t + st];
        __syncthreads();
    }
    const float inv = rsqrtf(red[0] * (1.0f / DM_) + 1e-5f);
    for (int i = t; i < DM_; i += 256) {
        float o = (xr[i] - mean) * inv * g[i] + b[i];
        yb[(long)blockIdx.x * DM_ + i] = (bf16)o;
        if (yf) yf[(long)blockIdx.x * DM_ + i] = o;
    }
}

// h += gate * LN(h, g, b) — one block per row, in place on h (f32)
__global__ __launch_bounds__(256) void gate_ln_update_kernel(float* __restrict__ h,
                                                             const float* __restrict__ gate,
                                                             const float* __restrict__ g,
                                                             const float* __restrict__ b) {
    __shared__ float red[256];
    const int t = threadIdx.x;
    float* xr = h + (long)blockIdx.x * DM_;
    const float* gr = gate + (long)blockIdx.x * DM_;
    float s = 0.f;
    for (int i = t; i < DM_; i += 256) s += xr[i];
    red[t] = s; __syncthreads();
    for (int st = 128; st > 0; st >>= 1) {
        if (t < st) red[t] += red[t + st];
        __syncthreads();
    }
    const float mean = red[0] * (1.0f / DM_);
    __syncthreads();
    float v = 0.f;
    for (int i = t; i < DM_; i += 256) { float d = xr[i] - mean; v += d * d; }
    red[t] = v; __syncthreads();
    for (int st = 128; st > 0; st >>= 1) {
        if (t < st) red[t] += red[t + st];
        __syncthreads();
    }
    const float inv = rsqrtf(red[0] * (1.0f / DM_) + 1e-5f);
    __syncthreads();
    for (int i = t; i < DM_; i += 256) {
        float xv = xr[i];
        float ln = (xv - mean) * inv * g[i] + b[i];
        xr[i] = xv + gr[i] * ln;
    }
}

// ---------------------------------------------------------------------------
// bf16 WMMA GEMM:  out[M,N] = epilogue( A[M,K] @ W^T[N,K] + bias[N] )
// block = 256 threads = 8 waves; block tile 256(M) x 64(N); wave tile 32x64.
// B tile (64 cols x 32 K) staged in LDS via global_load_async_to_lds_b128,
// double-buffered (ASYNCcnt); A fragments register-pipelined from global.
// Steady-state loop is branch-free (last K-step peeled); all 4 B fragments
// preloaded per step so LDS latency is paid once per step.
// EPI: 0 = none, 1 = exact GELU, 2 = += residual, 3 = sigmoid,
//      4 = outF[r,c] += coeff[r]*x   (MoE combine accumulate)
// ---------------------------------------------------------------------------
template <int EPI, bool OUTB>
__global__ __launch_bounds__(256) void gemm_bf16_kernel(
    const bf16* __restrict__ A, const bf16* __restrict__ Wt,
    const float* __restrict__ bias, const float* __restrict__ extra,
    float* __restrict__ outF, bf16* __restrict__ outB,
    int M, int N, int K, int coeffLD, int coeffIdx)
{
    __shared__ __align__(16) bf16 Bsm[2][64 * BSTRIDE];

    const int tid  = threadIdx.x;
    const int lane = tid & 31;
    const int wave = tid >> 5;
    const int hl   = lane >> 4;     // lane-half select (0/1)
    const int ll   = lane & 15;
    const int rowT = blockIdx.y * 256 + wave * 32;
    const int colT = blockIdx.x * 64;

    // --- async B-tile copy assignment: each thread moves one 16B chunk ------
    const int bcol = tid >> 2;            // 0..63  (tile-local column)
    const int bk   = (tid & 3) * 8;       // 0,8,16,24 (K offset)
    const bf16* bsrc = Wt + (long)(colT + bcol) * K + bk;
    const unsigned ldsBase =
        (unsigned)(size_t)(void*)(&Bsm[0][0]) + (unsigned)(bcol * BSTRIDE + bk) * 2u;

    // --- A fragment pointers (two 16-row fragments per wave) ----------------
    const bf16* Arow0 = A + (long)(rowT + ll) * K + hl * 8;
    const bf16* Arow1 = A + (long)(rowT + 16 + ll) * K + hl * 8;

    union U { uint4 q[2]; v16bf v; };
    v8f acc[2][4] = {};

    const int ksteps = K / 32;

    // prologue: stream tile 0 into buffer 0, prefetch A frags for step 0
    async_copy_b128(ldsBase, bsrc);
    U a0, a1;
    a0.q[0] = *(const uint4*)(Arow0);
    a0.q[1] = *(const uint4*)(Arow0 + 16);
    a1.q[0] = *(const uint4*)(Arow1);
    a1.q[1] = *(const uint4*)(Arow1 + 16);

    int buf = 0;
    // steady state: branch-free; last step peeled below
    for (int ks = 0; ks < ksteps - 1; ++ks) {
        // stream next B tile into the other buffer; prefetch next A frags
        async_copy_b128(ldsBase + (unsigned)((buf ^ 1) * BUFBYTES),
                        bsrc + (long)(ks + 1) * 32);
        const long ko = (long)(ks + 1) * 32;
        U n0, n1;
        n0.q[0] = *(const uint4*)(Arow0 + ko);
        n0.q[1] = *(const uint4*)(Arow0 + ko + 16);
        n1.q[0] = *(const uint4*)(Arow1 + ko);
        n1.q[1] = *(const uint4*)(Arow1 + ko + 16);

        wait_async_le1();              // current buffer's copy done
        __syncthreads();               // B tile visible to all waves

        U bfr[4];
#pragma unroll
        for (int tt = 0; tt < 4; ++tt) {
            const bf16* bp = &Bsm[buf][(tt * 16 + ll) * BSTRIDE + hl * 8];
            bfr[tt].q[0] = *(const uint4*)(bp);      // ds_load_b128 (clustered)
            bfr[tt].q[1] = *(const uint4*)(bp + 16);
        }
#pragma unroll
        for (int tt = 0; tt < 4; ++tt) {
            acc[0][tt] = __builtin_amdgcn_wmma_f32_16x16x32_bf16(
                false, a0.v, false, bfr[tt].v, (short)0, acc[0][tt], false, false);
            acc[1][tt] = __builtin_amdgcn_wmma_f32_16x16x32_bf16(
                false, a1.v, false, bfr[tt].v, (short)0, acc[1][tt], false, false);
        }
        __syncthreads();               // all reads done before buffer overwrite
        a0 = n0; a1 = n1;
        buf ^= 1;
    }

    // peeled last K-step
    wait_async_le0();
    __syncthreads();
    {
        U bfr[4];
#pragma unroll
        for (int tt = 0; tt < 4; ++tt) {
            const bf16* bp = &Bsm[buf][(tt * 16 + ll) * BSTRIDE + hl * 8];
            bfr[tt].q[0] = *(const uint4*)(bp);
            bfr[tt].q[1] = *(const uint4*)(bp + 16);
        }
#pragma unroll
        for (int tt = 0; tt < 4; ++tt) {
            acc[0][tt] = __builtin_amdgcn_wmma_f32_16x16x32_bf16(
                false, a0.v, false, bfr[tt].v, (short)0, acc[0][tt], false, false);
            acc[1][tt] = __builtin_amdgcn_wmma_f32_16x16x32_bf16(
                false, a1.v, false, bfr[tt].v, (short)0, acc[1][tt], false, false);
        }
    }

#pragma unroll
    for (int mt = 0; mt < 2; ++mt) {
#pragma unroll
        for (int tt = 0; tt < 4; ++tt) {
            const int col = colT + tt * 16 + ll;
            const float bval = bias ? bias[col] : 0.f;
#pragma unroll
            for (int vv = 0; vv < 8; ++vv) {
                const int row = rowT + mt * 16 + hl * 8 + vv;
                const long oi = (long)row * N + col;
                float x = acc[mt][tt][vv] + bval;
                if (EPI == 1) x = 0.5f * x * (1.0f + erff(x * 0.70710678118f));
                else if (EPI == 2) x += extra[oi];
                else if (EPI == 3) x = 1.0f / (1.0f + __expf(-x));
                if (EPI == 4) {
                    outF[oi] += extra[(long)row * coeffLD + coeffIdx] * x;
                } else if (OUTB) {
                    outB[oi] = (bf16)x;
                } else {
                    outF[oi] = x;
                }
            }
        }
    }
}

// ---------------------------------------------------------------------------
// Flash-style attention over qkv f32 [NT, 3*DM]; one wave per (b, q, h).
// ---------------------------------------------------------------------------
__global__ __launch_bounds__(256) void attn_kernel(const float* __restrict__ qkv,
                                                   bf16* __restrict__ ao) {
    const int lane = threadIdx.x & 31;
    const int wave = threadIdx.x >> 5;
    const long idx  = (long)blockIdx.x * 8 + wave;   // over B*S*H
    const int  h    = (int)(idx % H_);
    const long bq   = idx / H_;                      // token row = b*S + q
    const int  b    = (int)(bq / S_);

    const float* qp = qkv + bq * (3 * DM_) + h * HD_ + lane * 2;
    const float q0 = qp[0], q1 = qp[1];

    const float* kbase = qkv + (long)b * S_ * (3 * DM_) + DM_ + h * HD_ + lane * 2;
    const float* vbase = kbase + DM_;

    float m = -1e30f, l = 0.f, a0 = 0.f, a1 = 0.f;
    for (int kpos = 0; kpos < S_; ++kpos) {
        const float* kp = kbase + (long)kpos * (3 * DM_);
        float s = q0 * kp[0] + q1 * kp[1];
        for (int off = 16; off > 0; off >>= 1) s += __shfl_xor(s, off, 32);
        s *= 0.125f;                                 // 1/sqrt(64)
        const float mn = fmaxf(m, s);
        const float co = __expf(m - mn);
        const float p  = __expf(s - mn);
        const float* vp = vbase + (long)kpos * (3 * DM_);
        a0 = a0 * co + p * vp[0];
        a1 = a1 * co + p * vp[1];
        l  = l * co + p;
        m  = mn;
    }
    const float inv = 1.0f / l;
    bf16* op = ao + bq * DM_ + h * HD_ + lane * 2;
    op[0] = (bf16)(a0 * inv);
    op[1] = (bf16)(a1 * inv);
}

// ---------------------------------------------------------------------------
// Router: one block per token. accum[0..3]=selection counts, [4..7]=prob sums.
// ---------------------------------------------------------------------------
__global__ __launch_bounds__(256) void router_kernel(const float* __restrict__ h2,
                                                     const float* __restrict__ rw,
                                                     const float* __restrict__ rb,
                                                     float* __restrict__ combine,
                                                     float* __restrict__ accum) {
    __shared__ float red[E_][256];
    const int t = threadIdx.x;
    const float* x = h2 + (long)blockIdx.x * DM_;
    float p[E_] = {0.f, 0.f, 0.f, 0.f};
    for (int i = t; i < DM_; i += 256) {
        const float xv = x[i];
#pragma unroll
        for (int e = 0; e < E_; ++e) p[e] += xv * rw[i * E_ + e];
    }
#pragma unroll
    for (int e = 0; e < E_; ++e) red[e][t] = p[e];
    __syncthreads();
    for (int st = 128; st > 0; st >>= 1) {
        if (t < st)
#pragma unroll
            for (int e = 0; e < E_; ++e) red[e][t] += red[e][t + st];
        __syncthreads();
    }
    if (t == 0) {
        float lg[E_], pr[E_];
        float mx = -1e30f;
        for (int e = 0; e < E_; ++e) { lg[e] = red[e][0] + rb[e]; mx = fmaxf(mx, lg[e]); }
        float se = 0.f;
        for (int e = 0; e < E_; ++e) { pr[e] = __expf(lg[e] - mx); se += pr[e]; }
        const float ise = 1.0f / se;
        for (int e = 0; e < E_; ++e) pr[e] *= ise;
        int i0 = 0;
        for (int e = 1; e < E_; ++e) if (pr[e] > pr[i0]) i0 = e;
        int i1 = -1;
        for (int e = 0; e < E_; ++e)
            if (e != i0 && (i1 < 0 || pr[e] > pr[i1])) i1 = e;
        const float wsum = pr[i0] + pr[i1];
        float c[E_] = {0.f, 0.f, 0.f, 0.f};
        c[i0] = pr[i0] / wsum;
        c[i1] = pr[i1] / wsum;
        for (int e = 0; e < E_; ++e) combine[(long)blockIdx.x * E_ + e] = c[e];
        atomicAdd(&accum[i0], 1.0f);
        atomicAdd(&accum[i1], 1.0f);
        for (int e = 0; e < E_; ++e) atomicAdd(&accum[E_ + e], pr[e]);
    }
}

__global__ void finalize_aux_kernel(const float* __restrict__ accum, float* __restrict__ out_aux) {
    float s = 0.f;
    for (int e = 0; e < E_; ++e)
        s += (accum[e] / (float)(NT_ * KTOP_)) * (accum[E_ + e] / (float)NT_);
    *out_aux = (float)E_ * s;
}

// ---------------------------------------------------------------------------
// Host launcher
// ---------------------------------------------------------------------------
static inline int gsblocks(long n) {
    long b = (n + 255) / 256;
    return (int)(b > 16384 ? 16384 : b);
}

extern "C" void kernel_launch(void* const* d_in, const int* in_sizes, int n_in,
                              void* d_out, int out_size, void* d_ws, size_t ws_size,
                              hipStream_t stream) {
    const float* hidden   = (const float*)d_in[0];
    const float* qkv_w    = (const float*)d_in[1];
    const float* qkv_b    = (const float*)d_in[2];
    const float* out_w    = (const float*)d_in[3];
    const float* out_b    = (const float*)d_in[4];
    const float* gate_w   = (const float*)d_in[5];
    const float* gate_b   = (const float*)d_in[6];
    const float* inorm_g  = (const float*)d_in[7];
    const float* inorm_b  = (const float*)d_in[8];
    const float* anorm_g  = (const float*)d_in[9];
    const float* anorm_b  = (const float*)d_in[10];
    const float* fnorm_g  = (const float*)d_in[11];
    const float* fnorm_b  = (const float*)d_in[12];
    const float* router_w = (const float*)d_in[13];
    const float* router_b = (const float*)d_in[14];
    const float* e_w1     = (const float*)d_in[15];
    const float* e_b1     = (const float*)d_in[16];
    const float* e_w2     = (const float*)d_in[17];
    const float* e_b2     = (const float*)d_in[18];
    float* out = (float*)d_out;

    // ---- workspace carve-up -------------------------------------------------
    char* ws = (char*)d_ws;
    auto carve = [&](size_t bytes) -> char* {
        char* p = ws;
        ws += (bytes + 255) & ~(size_t)255;
        return p;
    };
    bf16*  wt_qkv  = (bf16*)carve((size_t)DM_ * 3 * DM_ * sizeof(bf16));   // [3DM, DM]
    bf16*  wt_out  = (bf16*)carve((size_t)DM_ * DM_ * sizeof(bf16));       // [DM, DM]
    bf16*  wt_gate = (bf16*)carve((size_t)DM_ * DM_ * sizeof(bf16));       // [DM, DM]
    bf16*  wt_e1   = (bf16*)carve((size_t)E_ * DM_ * FF_ * sizeof(bf16));  // E x [FF, DM]
    bf16*  wt_e2   = (bf16*)carve((size_t)E_ * FF_ * DM_ * sizeof(bf16));  // E x [DM, FF]
    bf16*  ln1_b   = (bf16*)carve((size_t)NT_ * DM_ * sizeof(bf16));
    float* qkv_f   = (float*)carve((size_t)NT_ * 3 * DM_ * sizeof(float));
    bf16*  ao_b    = (bf16*)carve((size_t)NT_ * DM_ * sizeof(bf16));
    float* h_f     = (float*)carve((size_t)NT_ * DM_ * sizeof(float));
    bf16*  h_b     = (bf16*)carve((size_t)NT_ * DM_ * sizeof(bf16));
    float* gate_f  = (float*)carve((size_t)NT_ * DM_ * sizeof(float));
    float* h2_f    = (float*)carve((size_t)NT_ * DM_ * sizeof(float));
    bf16*  h2_b    = (bf16*)carve((size_t)NT_ * DM_ * sizeof(bf16));
    bf16*  act_b   = (bf16*)carve((size_t)NT_ * FF_ * sizeof(bf16));
    float* combine = (float*)carve((size_t)NT_ * E_ * sizeof(float));
    float* accum   = (float*)carve(2 * E_ * sizeof(float));

    // ---- weight conversion (f32 -> transposed bf16) ------------------------
    transpose_cvt_kernel<<<gsblocks((long)DM_ * 3 * DM_), 256, 0, stream>>>(qkv_w,  wt_qkv,  DM_, 3 * DM_);
    transpose_cvt_kernel<<<gsblocks((long)DM_ * DM_),     256, 0, stream>>>(out_w,  wt_out,  DM_, DM_);
    transpose_cvt_kernel<<<gsblocks((long)DM_ * DM_),     256, 0, stream>>>(gate_w, wt_gate, DM_, DM_);
    for (int e = 0; e < E_; ++e) {
        transpose_cvt_kernel<<<gsblocks((long)DM_ * FF_), 256, 0, stream>>>(
            e_w1 + (size_t)e * DM_ * FF_, wt_e1 + (size_t)e * FF_ * DM_, DM_, FF_);
        transpose_cvt_kernel<<<gsblocks((long)FF_ * DM_), 256, 0, stream>>>(
            e_w2 + (size_t)e * FF_ * DM_, wt_e2 + (size_t)e * DM_ * FF_, FF_, DM_);
    }

    // ---- attention block ----------------------------------------------------
    ln_kernel<<<NT_, 256, 0, stream>>>(hidden, anorm_g, anorm_b, ln1_b, nullptr);

    gemm_bf16_kernel<0, false><<<dim3(3 * DM_ / 64, NT_ / 256), 256, 0, stream>>>(
        ln1_b, wt_qkv, qkv_b, nullptr, qkv_f, nullptr, NT_, 3 * DM_, DM_, 0, 0);

    attn_kernel<<<NT_ * H_ / 8, 256, 0, stream>>>(qkv_f, ao_b);

    gemm_bf16_kernel<2, false><<<dim3(DM_ / 64, NT_ / 256), 256, 0, stream>>>(
        ao_b, wt_out, out_b, hidden, h_f, nullptr, NT_, DM_, DM_, 0, 0);

    // ---- gated-LN iterations ------------------------------------------------
    for (int it = 0; it < IT_; ++it) {
        cvt_bf16_kernel<<<gsblocks((long)NT_ * DM_), 256, 0, stream>>>(h_f, h_b, (long)NT_ * DM_);
        gemm_bf16_kernel<3, false><<<dim3(DM_ / 64, NT_ / 256), 256, 0, stream>>>(
            h_b, wt_gate, gate_b, nullptr, gate_f, nullptr, NT_, DM_, DM_, 0, 0);
        gate_ln_update_kernel<<<NT_, 256, 0, stream>>>(h_f, gate_f, inorm_g, inorm_b);
    }

    // ---- router + MoE -------------------------------------------------------
    ln_kernel<<<NT_, 256, 0, stream>>>(h_f, fnorm_g, fnorm_b, h2_b, h2_f);

    hipMemsetAsync(accum, 0, 2 * E_ * sizeof(float), stream);
    router_kernel<<<NT_, 256, 0, stream>>>(h2_f, router_w, router_b, combine, accum);

    // out = residual (h), experts accumulate into it
    copy_f32_kernel<<<gsblocks((long)NT_ * DM_), 256, 0, stream>>>(h_f, out, (long)NT_ * DM_);

    for (int e = 0; e < E_; ++e) {
        gemm_bf16_kernel<1, true><<<dim3(FF_ / 64, NT_ / 256), 256, 0, stream>>>(
            h2_b, wt_e1 + (size_t)e * FF_ * DM_, e_b1 + (size_t)e * FF_, nullptr,
            nullptr, act_b, NT_, FF_, DM_, 0, 0);
        gemm_bf16_kernel<4, false><<<dim3(DM_ / 64, NT_ / 256), 256, 0, stream>>>(
            act_b, wt_e2 + (size_t)e * DM_ * FF_, e_b2 + (size_t)e * DM_, combine,
            out, nullptr, NT_, DM_, FF_, E_, e);
    }

    finalize_aux_kernel<<<1, 1, 0, stream>>>(accum, out + (size_t)NT_ * DM_);
}
